// TcLstm_50122268345039
// MI455X (gfx1250) — matrix operand order repaired and checked by reading, not compiled, not run on hardware
//
#include <hip/hip_runtime.h>
#include <hip/hip_bf16.h>
#include <cstdint>

// Problem constants (from reference)
#define BB   256
#define LL   128
#define TTT  8
#define VV   50000
#define DD   300
#define HH   200
#define CC   3
#define N4H  800      // 4*H gate width
#define DK   320      // D padded to multiple of 32 for WMMA K
#define KT_E 10       // DK/32 K-chunks for e-part GEMM
#define HK   224      // H padded to multiple of 32
#define KT_H 7        // HK/32 K-chunks for recurrent GEMM
#define NT_E 50       // N4H/16 column tiles (input GEMM packing)
#define NT_HP 64      // recurrent packing padded to 4 tiles x 16 waves

typedef __attribute__((ext_vector_type(16))) _Float16 v16h;
typedef __attribute__((ext_vector_type(8)))  float    v8f;

union V16U { v16h v; uint4 q[2]; };

// A-operand load (16x32 f16 tile, ISA 7.12.2): lane l (m = l&15) holds
// halves 0..7 -> K = k0..k0+7, halves 8..15 -> K = k0+16..k0+23,
// k0 = ktBase + (l<16 ? 0 : 8). p -> (row m, half k0), 16B aligned.
__device__ __forceinline__ v16h ldA(const _Float16* p) {
  V16U u;
  u.q[0] = *reinterpret_cast<const uint4*>(p);
  u.q[1] = *reinterpret_cast<const uint4*>(p + 16);
  return u.v;
}

// B-operand load from pre-swizzled buffer: 32 contiguous bytes per lane.
__device__ __forceinline__ v16h ldB(const _Float16* p) {
  V16U u;
  const uint4* q = reinterpret_cast<const uint4*>(p);
  u.q[0] = q[0];
  u.q[1] = q[1];
  return u.v;
}

__device__ __forceinline__ v8f wmma16(v16h a, v16h b, v8f c) {
  return __builtin_amdgcn_wmma_f32_16x16x32_f16(false, a, false, b,
                                                (short)0, c, false, false);
}

__device__ __forceinline__ float sigf(float x) { return 1.0f / (1.0f + expf(-x)); }

// ---------------------------------------------------------------- prep kernels

// emb f32 [V,300] -> f16 [V,320] zero-padded (rows 640B, 16B aligned)
__global__ void k_embh(const float* __restrict__ emb, _Float16* __restrict__ embh) {
  int row = blockIdx.x, d = threadIdx.x;
  embh[(size_t)row * DK + d] = (d < DD) ? (_Float16)emb[(size_t)row * DD + d]
                                        : (_Float16)0.0f;
}

// masked mean of target embeddings -> mean[256,300] f32
__global__ void k_mean(const int* __restrict__ tids, const int* __restrict__ tlen,
                       const float* __restrict__ emb, float* __restrict__ mean) {
  int b = blockIdx.x, d = threadIdx.x;
  if (d >= DD) return;
  int n = tlen[b];
  if (n > TTT) n = TTT;
  float s = 0.0f;
  for (int t = 0; t < n; ++t)
    s += emb[(size_t)tids[b * TTT + t] * DD + d];
  mean[b * DD + d] = s / (float)tlen[b];
}

// zc[b,n] = bias[n] + mean[b,:] @ W[300:600, n]   (f32, tiny GEMM)
__global__ void k_zc(const float* __restrict__ mean, const float* __restrict__ W,
                     const float* __restrict__ bias, float* __restrict__ zc) {
  int idx = blockIdx.x * blockDim.x + threadIdx.x;   // over 256*800
  int b = idx / N4H, n = idx % N4H;
  float s = bias[n];
  const float* mp = mean + b * DD;
  const float* wp = W + (size_t)DD * N4H + n;        // rows 300..599
  for (int d = 0; d < DD; ++d) s += mp[d] * wp[(size_t)d * N4H];
  zc[idx] = s;
}

// Pack a K-slab of W (f32 [800,800]) into WMMA B-operand lane layout.
// WB[((kt*NTP+nt)*32+lane)*16+h] =
//   W[rowOff + kt*32 + (lane<16?0:8) + (h<8?h:h+8)][nt*16 + (lane&15)], zero-padded.
__global__ void k_packW(const float* __restrict__ W, _Float16* __restrict__ WB,
                        int KT, int NTP, int Krows, int rowOff) {
  int idx = blockIdx.x * blockDim.x + threadIdx.x;
  if (idx >= KT * NTP * 32 * 16) return;
  int h = idx & 15;
  int lane = (idx >> 4) & 31;
  int rest = idx >> 9;
  int nt = rest % NTP;
  int kt = rest / NTP;
  int k = kt * 32 + ((lane < 16) ? 0 : 8) + ((h < 8) ? h : h + 8);
  int col = nt * 16 + (lane & 15);
  float v = (k < Krows && col < N4H) ? W[(size_t)(rowOff + k) * N4H + col] : 0.0f;
  WB[idx] = (_Float16)v;
}

// ------------------------------------------------- time-parallel input GEMM
// P[row, n] = embh[ids[row], :] @ W_e + zc[row/L, n]   (zc folded in here!)
// One wave per block, M=32 (two row tiles reuse each B fetch), 5 column tiles.
// Explicit double-buffer of A and B across kt so loads overlap WMMAs.
// grid = (1024, 10).
__global__ void __launch_bounds__(32)
k_xproj(const int* __restrict__ ids, const _Float16* __restrict__ embh,
        const _Float16* __restrict__ WB, const float* __restrict__ zc,
        _Float16* __restrict__ P) {
  const int lane = threadIdx.x;
  const int rowBase = blockIdx.x * 32;
  const int b = rowBase / LL;                   // uniform: 32 | 128
  const int m = lane & 15;
  const int hi = (lane < 16) ? 0 : 8;
  const _Float16* rp0 = embh + (size_t)ids[rowBase + m] * DK;
  const _Float16* rp1 = embh + (size_t)ids[rowBase + 16 + m] * DK;

  v8f acc[2][5];
#pragma unroll
  for (int rt = 0; rt < 2; ++rt)
#pragma unroll
    for (int i = 0; i < 5; ++i)
      acc[rt][i] = (v8f){0.f,0.f,0.f,0.f,0.f,0.f,0.f,0.f};

  // stage 0 operands
  v16h a0c = ldA(rp0 + hi);
  v16h a1c = ldA(rp1 + hi);
  v16h bcur[5];
#pragma unroll
  for (int i = 0; i < 5; ++i) {
    int nt = blockIdx.y * 5 + i;
    bcur[i] = ldB(WB + ((size_t)(0 * NT_E + nt) * 32 + lane) * 16);
  }

  for (int kt = 0; kt < KT_E; ++kt) {
    const int kn = (kt < KT_E - 1) ? kt + 1 : kt;   // prefetch next stage
    v16h a0n = ldA(rp0 + kn * 32 + hi);
    v16h a1n = ldA(rp1 + kn * 32 + hi);
    v16h bnxt[5];
#pragma unroll
    for (int i = 0; i < 5; ++i) {
      int nt = blockIdx.y * 5 + i;
      bnxt[i] = ldB(WB + ((size_t)(kn * NT_E + nt) * 32 + lane) * 16);
    }
#pragma unroll
    for (int i = 0; i < 5; ++i) acc[0][i] = wmma16(a0c, bcur[i], acc[0][i]);
#pragma unroll
    for (int i = 0; i < 5; ++i) acc[1][i] = wmma16(a1c, bcur[i], acc[1][i]);
    a0c = a0n; a1c = a1n;
#pragma unroll
    for (int i = 0; i < 5; ++i) bcur[i] = bnxt[i];
  }

#pragma unroll
  for (int rt = 0; rt < 2; ++rt)
#pragma unroll
    for (int i = 0; i < 5; ++i) {
      const int col = (blockIdx.y * 5 + i) * 16 + m;
      const float zcv = zc[(size_t)b * N4H + col];
#pragma unroll
      for (int v = 0; v < 8; ++v) {
        const int row = rowBase + rt * 16 + v + hi;  // C layout: M = v + (lane<16?0:8)
        P[(size_t)row * N4H + col] = (_Float16)(acc[rt][i][v] + zcv);
      }
    }
}

// ------------------------------------------------------- recurrent LSTM tiles
// grid (16, 2): 16 batch-tiles x {left,right}. 512 threads = 16 waves.
// W_h is REGISTER-RESIDENT: each wave owns 4 column tiles x 7 K-chunks
// (Breg = 224 VGPRs), loaded once; the 128-step loop touches only LDS (h in
// A-operand layout) and the 16x800 f16 P[t] tile. P loads are issued before
// each 7-deep WMMA chain so their latency hides behind the matrix ops.
__global__ void __launch_bounds__(512, 1)
k_lstm(const int* __restrict__ lenL, const int* __restrict__ lenR,
       const _Float16* __restrict__ PL, const _Float16* __restrict__ PR,
       const _Float16* __restrict__ WhBL, const _Float16* __restrict__ WhBR,
       float* __restrict__ HoutL, float* __restrict__ HoutR) {
  const int b0 = blockIdx.x * 16;
  const int side = blockIdx.y;
  const int* len       = side ? lenR  : lenL;
  const _Float16* P    = side ? PR    : PL;
  const _Float16* WB   = side ? WhBR  : WhBL;
  float* Hout          = side ? HoutR : HoutL;

  extern __shared__ char smem[];
  _Float16* hbuf = (_Float16*)smem;                        // 16 x 224 f16
  float* zbuf = (float*)(smem + 16 * HK * 2);              // 16 x 800 f32
  float* cbuf = zbuf + 16 * N4H;                           // 16 x 200 f32
  float* hf   = cbuf + 16 * HH;                            // 16 x 200 f32

  const int tid = threadIdx.x;
  const int w = tid >> 5;
  const int lane = tid & 31;
  const int m = lane & 15;
  const int hi = (lane < 16) ? 0 : 8;

  // One-time load of this wave's W_h tiles into registers (padded to nt<64).
  v16h Breg[4][7];
#pragma unroll
  for (int i = 0; i < 4; ++i) {
    const int nt = w * 4 + i;
#pragma unroll
    for (int kt = 0; kt < KT_H; ++kt)
      Breg[i][kt] = ldB(WB + ((size_t)(kt * NT_HP + nt) * 32 + lane) * 16);
  }

  for (int i = tid; i < 16 * HK; i += 512) hbuf[i] = (_Float16)0.0f;
  for (int i = tid; i < 16 * HH; i += 512) { cbuf[i] = 0.0f; hf[i] = 0.0f; }
  __syncthreads();

  for (int t = 0; t < LL; ++t) {
#pragma unroll
    for (int i = 0; i < 4; ++i) {
      const int nt = w * 4 + i;
      if (nt < 50) {                        // wave-uniform guard
        const int col = nt * 16 + m;
        // issue P loads first: they retire behind the 7-deep WMMA chain
        float pv[8];
#pragma unroll
        for (int v = 0; v < 8; ++v)
          pv[v] = (float)P[((size_t)(b0 + v + hi) * LL + t) * N4H + col];
        v8f acc = (v8f){0.f,0.f,0.f,0.f,0.f,0.f,0.f,0.f};
#pragma unroll
        for (int kt = 0; kt < KT_H; ++kt) {
          v16h a = ldA(hbuf + m * HK + kt * 32 + hi);       // ds_load_b128 x2
          acc = wmma16(a, Breg[i][kt], acc);
        }
#pragma unroll
        for (int v = 0; v < 8; ++v)
          zbuf[(v + hi) * N4H + col] = acc[v] + pv[v];
      }
    }
    __syncthreads();

    // gates + masked state update (z split into i,j,f,o of width H)
    for (int i2 = tid; i2 < 16 * HH; i2 += 512) {
      const int r = i2 / HH, j = i2 % HH;
      if (t < len[b0 + r]) {
        const float zi = zbuf[r * N4H + j];
        const float zj = zbuf[r * N4H + HH + j];
        const float zf = zbuf[r * N4H + 2 * HH + j];
        const float zo = zbuf[r * N4H + 3 * HH + j];
        const float cn = sigf(zf + 1.0f) * cbuf[i2] + sigf(zi) * tanhf(zj);
        const float hn = sigf(zo) * tanhf(cn);
        cbuf[i2] = cn;
        hf[i2] = hn;
        hbuf[r * HK + j] = (_Float16)hn;    // padding halves [200,224) stay 0
      }
    }
    __syncthreads();
  }

  for (int i2 = tid; i2 < 16 * HH; i2 += 512)
    Hout[(size_t)b0 * HH + i2] = hf[i2];
}

// ------------------------------------------------------------------- decoder
__global__ void __launch_bounds__(64)
k_logits(const float* __restrict__ Hl, const float* __restrict__ Hr,
         const float* __restrict__ Wd, const float* __restrict__ bd,
         float* __restrict__ out) {
  __shared__ float red[3][64];
  const int b = blockIdx.x, tid = threadIdx.x;
  float p0 = 0.f, p1 = 0.f, p2 = 0.f;
  for (int j = tid; j < 2 * HH; j += 64) {
    const float hv = (j < HH) ? Hl[b * HH + j] : Hr[b * HH + (j - HH)];
    p0 += hv * Wd[j * CC + 0];
    p1 += hv * Wd[j * CC + 1];
    p2 += hv * Wd[j * CC + 2];
  }
  red[0][tid] = p0; red[1][tid] = p1; red[2][tid] = p2;
  __syncthreads();
  for (int s = 32; s > 0; s >>= 1) {
    if (tid < s) {
      red[0][tid] += red[0][tid + s];
      red[1][tid] += red[1][tid + s];
      red[2][tid] += red[2][tid + s];
    }
    __syncthreads();
  }
  if (tid < CC) out[b * CC + tid] = red[tid][0] + bd[tid];
}

// ------------------------------------------------------------------- launcher
extern "C" void kernel_launch(void* const* d_in, const int* in_sizes, int n_in,
                              void* d_out, int out_size, void* d_ws, size_t ws_size,
                              hipStream_t stream) {
  const int*   left_ids   = (const int*)d_in[0];
  const int*   target_ids = (const int*)d_in[1];
  const int*   right_ids  = (const int*)d_in[2];
  const int*   left_len   = (const int*)d_in[3];
  const int*   target_len = (const int*)d_in[4];
  const int*   right_len  = (const int*)d_in[5];
  const float* emb        = (const float*)d_in[6];
  const float* Wl         = (const float*)d_in[7];
  const float* bl         = (const float*)d_in[8];
  const float* Wr         = (const float*)d_in[9];
  const float* br         = (const float*)d_in[10];
  const float* Wd         = (const float*)d_in[11];
  const float* bd         = (const float*)d_in[12];
  float* out = (float*)d_out;

  // Workspace carve-up (512B aligned slices). Total ~140 MB.
  char* ws = (char*)d_ws;
  size_t off = 0;
  auto alloc = [&](size_t bytes) -> size_t {
    size_t o = off;
    off += (bytes + 511) & ~(size_t)511;
    return o;
  };
  const size_t o_embh  = alloc((size_t)VV * DK * 2);              // 32.0 MB f16
  const size_t o_mean  = alloc((size_t)BB * DD * 4);
  const size_t o_zcL   = alloc((size_t)BB * N4H * 4);
  const size_t o_zcR   = alloc((size_t)BB * N4H * 4);
  const size_t o_WeBL  = alloc((size_t)KT_E * NT_E * 32 * 16 * 2);
  const size_t o_WeBR  = alloc((size_t)KT_E * NT_E * 32 * 16 * 2);
  const size_t o_WhBL  = alloc((size_t)KT_H * NT_HP * 32 * 16 * 2);
  const size_t o_WhBR  = alloc((size_t)KT_H * NT_HP * 32 * 16 * 2);
  const size_t o_PL    = alloc((size_t)BB * LL * N4H * 2);        // 52.4 MB f16
  const size_t o_PR    = alloc((size_t)BB * LL * N4H * 2);
  const size_t o_HoutL = alloc((size_t)BB * HH * 4);
  const size_t o_HoutR = alloc((size_t)BB * HH * 4);
  (void)ws_size; (void)in_sizes; (void)n_in; (void)out_size;

  _Float16* embh = (_Float16*)(ws + o_embh);
  float*    mean = (float*)(ws + o_mean);
  float*    zcL  = (float*)(ws + o_zcL);
  float*    zcR  = (float*)(ws + o_zcR);
  _Float16* WeBL = (_Float16*)(ws + o_WeBL);
  _Float16* WeBR = (_Float16*)(ws + o_WeBR);
  _Float16* WhBL = (_Float16*)(ws + o_WhBL);
  _Float16* WhBR = (_Float16*)(ws + o_WhBR);
  _Float16* PL   = (_Float16*)(ws + o_PL);
  _Float16* PR   = (_Float16*)(ws + o_PR);
  float*    HoL  = (float*)(ws + o_HoutL);
  float*    HoR  = (float*)(ws + o_HoutR);

  // 1) f16 padded embedding table + target mean
  k_embh<<<VV, DK, 0, stream>>>(emb, embh);
  k_mean<<<BB, DK, 0, stream>>>(target_ids, target_len, emb, mean);

  // 2) time-invariant gate contribution zc = mean @ W_mid + bias (folded into P)
  k_zc<<<(BB * N4H) / 256, 256, 0, stream>>>(mean, Wl, bl, zcL);
  k_zc<<<(BB * N4H) / 256, 256, 0, stream>>>(mean, Wr, br, zcR);

  // 3) pack W_e (rows 0..299) and W_h (rows 600..799) into B-operand layout
  const int gE = (KT_E * NT_E  * 32 * 16 + 255) / 256;
  const int gH = (KT_H * NT_HP * 32 * 16 + 255) / 256;
  k_packW<<<gE, 256, 0, stream>>>(Wl, WeBL, KT_E, NT_E,  DD, 0);
  k_packW<<<gE, 256, 0, stream>>>(Wr, WeBR, KT_E, NT_E,  DD, 0);
  k_packW<<<gH, 256, 0, stream>>>(Wl, WhBL, KT_H, NT_HP, HH, 2 * DD);
  k_packW<<<gH, 256, 0, stream>>>(Wr, WhBR, KT_H, NT_HP, HH, 2 * DD);

  // 4) time-parallel input projections P = gather(embh, ids) @ W_e + zc (WMMA)
  dim3 gx(BB * LL / 32, NT_E / 5);
  k_xproj<<<gx, 32, 0, stream>>>(left_ids,  embh, WeBL, zcL, PL);
  k_xproj<<<gx, 32, 0, stream>>>(right_ids, embh, WeBR, zcR, PR);

  // 5) recurrent phase: 16 batch-tiles x 2 sides, register-resident W_h
  const size_t smem = (size_t)16 * HK * 2      // hbuf f16
                    + (size_t)16 * N4H * 4     // zbuf f32
                    + (size_t)16 * HH * 4 * 2; // cbuf + hf
  k_lstm<<<dim3(BB / 16, 2), 512, smem, stream>>>(left_len, right_len,
                                                  PL, PR, WhBL, WhBR, HoL, HoR);

  // 6) decoder
  k_logits<<<BB, 64, 0, stream>>>(HoL, HoR, Wd, bd, out);
}
